// MTCNN_36275293782255
// MI455X (gfx1250) — matrix operand order repaired and checked by reading, not compiled, not run on hardware
//
#include <hip/hip_runtime.h>
#include <math.h>

typedef _Float16 f16;
typedef __attribute__((ext_vector_type(8)))  _Float16 v8h;
typedef __attribute__((ext_vector_type(16))) _Float16 v16h;
typedef __attribute__((ext_vector_type(8)))  float    v8f;

#define NIMG 8

// ---------------------------------------------------------------------------
// Antialiased linear resize (matches jax.image.resize 'linear', antialias=True)
// horizontal pass: images NHWC (8,768,768,3) f32 -> tmp [img][c][768][nw] f32
// ---------------------------------------------------------------------------
__global__ __launch_bounds__(256) void k_hresize(const float* __restrict__ images,
                                                 float* __restrict__ tmp,
                                                 int nw, float s)
{
    int i = blockIdx.x * 256 + threadIdx.x;
    int tot = NIMG * 3 * 768 * nw;
    if (i >= tot) return;
    int xo = i % nw; int t = i / nw;
    int y  = t % 768; t /= 768;
    int c  = t % 3;   int img = t / 3;
    float c0 = (xo + 0.5f) / s - 0.5f;
    float ks = fminf(1.0f, s);
    float sup = 1.0f / ks;
    int j0 = (int)ceilf(c0 - sup); if (j0 < 0) j0 = 0;
    int j1 = (int)floorf(c0 + sup); if (j1 > 767) j1 = 767;
    float acc = 0.0f, wsum = 0.0f;
    for (int j = j0; j <= j1; ++j) {
        float wt = 1.0f - fabsf((float)j - c0) * ks;
        if (wt <= 0.0f) continue;
        wsum += wt;
        acc += wt * images[((img * 768 + y) * 768 + j) * 3 + c];
    }
    tmp[i] = (wsum > 0.0f) ? acc / wsum : 0.0f;
}

// vertical pass + normalize: tmp -> im f16 planar [img][c][nh][nw]
__global__ __launch_bounds__(256) void k_vresize(const float* __restrict__ tmp,
                                                 f16* __restrict__ im,
                                                 int nh, int nw, float s)
{
    int i = blockIdx.x * 256 + threadIdx.x;
    int tot = NIMG * 3 * nh * nw;
    if (i >= tot) return;
    int xo = i % nw; int t = i / nw;
    int yo = t % nh; t /= nh;
    int c  = t % 3;  int img = t / 3;
    float c0 = (yo + 0.5f) / s - 0.5f;
    float ks = fminf(1.0f, s);
    float sup = 1.0f / ks;
    int j0 = (int)ceilf(c0 - sup); if (j0 < 0) j0 = 0;
    int j1 = (int)floorf(c0 + sup); if (j1 > 767) j1 = 767;
    float acc = 0.0f, wsum = 0.0f;
    for (int j = j0; j <= j1; ++j) {
        float wt = 1.0f - fabsf((float)j - c0) * ks;
        if (wt <= 0.0f) continue;
        wsum += wt;
        acc += wt * tmp[((img * 3 + c) * 768 + j) * nw + xo];
    }
    float v = (wsum > 0.0f) ? acc / wsum : 0.0f;
    v = (v - 127.5f) * 0.0078125f;
    im[((img * 3 + c) * nh + yo) * nw + xo] = (f16)v;
}

// ---------------------------------------------------------------------------
// Generic 3x3 VALID conv + bias + PReLU via WMMA implicit GEMM.
//   in : f16 planar [8][Cin][H][W],  out: f16 planar [8][Cout][H-2][W-2]
//   NT = number of 16-wide output-channel tiles, KPAD = Cin*9 padded to 32.
// One wave computes one 16-pixel x (NT*16)-channel tile. 8 waves / block.
// A and B fragments use contiguous k order (k = 16*half + s); C/D per 16x16 f32 map.
// B tile stored TRANSPOSED in LDS ([n][k]) so a lane's fragment is 16
// contiguous halves -> ds_load_b128 pairs instead of 16 scalar reads.
// ---------------------------------------------------------------------------
template <int NT, int KPAD>
__global__ __launch_bounds__(256) void k_conv3x3_wmma(const f16* __restrict__ in,
                                                      f16* __restrict__ out,
                                                      const float* __restrict__ w,
                                                      const float* __restrict__ bias,
                                                      const float* __restrict__ prelu,
                                                      int Cin, int Cout, int H, int W)
{
    __shared__ f16 Bs[NT * 16 * KPAD];      // [n][k]
    __shared__ f16 As[8][16 * KPAD];        // per-wave [m][k]

    const int tid = threadIdx.x, lane = tid & 31, wv = tid >> 5;
    const int K = Cin * 9, NW = NT * 16;
    const int oH = H - 2, oW = W - 2;
    const int P = oH * oW;
    const int tilesPerImg = (P + 15) / 16;
    const int totalTiles = NIMG * tilesPerImg;

    // Weights -> LDS B tile, transposed, zero padded.
    for (int i = tid; i < NW * KPAD; i += 256) {
        int n = i / KPAD, k = i - n * KPAD;
        float v = 0.0f;
        if (k < K && n < Cout) v = w[(n * Cin + k / 9) * 9 + (k % 9)];
        Bs[i] = (f16)v;
    }

    const int half = lane >> 4, n0 = lane & 15, mrow = lane & 15;

    // Hoist bias / prelu coefficients (one channel per lane per N-tile);
    // issue these global loads early so they overlap the LDS fill.
    float bco[NT], pco[NT];
#pragma unroll
    for (int t = 0; t < NT; ++t) {
        int co = t * 16 + n0;
        int cc = co < Cout ? co : (Cout - 1);
        bco[t] = bias[cc];
        pco[t] = prelu[cc];
    }

    int tileG = blockIdx.x * 8 + wv;
    bool tv = tileG < totalTiles;
    int img = tv ? tileG / tilesPerImg : 0;
    int base = tv ? (tileG - img * tilesPerImg) * 16 : 0;
    const f16* inb = in + (size_t)img * Cin * H * W;
    __builtin_prefetch(inb, 0, 1);

    // im2col into per-wave LDS A tile [16][KPAD]
    f16* A = &As[wv][0];
    for (int m = 0; m < 16; ++m) {
        int p = base + m;
        bool pv = tv && (p < P);
        int y = pv ? p / oW : 0;
        int x = pv ? p - y * oW : 0;
        for (int k = lane; k < KPAD; k += 32) {
            float v = 0.0f;
            if (pv && k < K) {
                int ci = k / 9, r = k - ci * 9, kh = r / 3, kw = r - kh * 3;
                v = (float)inb[((size_t)ci * H + (y + kh)) * W + (x + kw)];
            }
            A[m * KPAD + k] = (f16)v;
        }
    }
    __syncthreads();

    v8f acc[NT];
#pragma unroll
    for (int t = 0; t < NT; ++t) { v8f z = {}; acc[t] = z; }

    const f16* Arow = &A[mrow * KPAD];

#pragma unroll
    for (int kc = 0; kc < KPAD / 32; ++kc) {
        const int kb = kc * 32;
        // A fragment: two contiguous 8-half runs (16B aligned each)
        v8h alo = *(const v8h*)&Arow[kb + half * 16];
        v8h ahi = *(const v8h*)&Arow[kb + half * 16 + 8];
        v16h a = __builtin_shufflevector(alo, ahi,
                                         0, 1, 2, 3, 4, 5, 6, 7,
                                         8, 9, 10, 11, 12, 13, 14, 15);
#pragma unroll
        for (int t = 0; t < NT; ++t) {
            // B fragment: 16 contiguous halves (32B aligned) in transposed tile
            v16h b = *(const v16h*)&Bs[(t * 16 + n0) * KPAD + kb + half * 16];
            acc[t] = __builtin_amdgcn_wmma_f32_16x16x32_f16(false, a, false, b,
                                                            (short)0, acc[t], false, false);
        }
    }

    // writeback: vgpr i holds M = i + 8*half, N = lane&15. Branchless PReLU.
    f16* outb = out + (size_t)img * Cout * P;
#pragma unroll
    for (int t = 0; t < NT; ++t) {
        int co = t * 16 + n0;
        bool cok = co < Cout;
#pragma unroll
        for (int i = 0; i < 8; ++i) {
            int p = base + i + half * 8;
            float v = acc[t][i] + bco[t];
            v = fmaxf(v, 0.0f) + pco[t] * fminf(v, 0.0f);
            if (tv && cok && p < P)
                outb[(size_t)co * P + p] = (f16)v;
        }
    }
}

// ---------------------------------------------------------------------------
// 2x2 stride-2 maxpool with end-padding to ceil (matches reduce_window pad H%2)
// ---------------------------------------------------------------------------
__global__ __launch_bounds__(256) void k_maxpool2(const f16* __restrict__ in,
                                                  f16* __restrict__ out,
                                                  int C, int h, int w, int ph, int pw)
{
    int i = blockIdx.x * 256 + threadIdx.x;
    int tot = NIMG * C * ph * pw;
    if (i >= tot) return;
    int x = i % pw; int t = i / pw;
    int y = t % ph; t /= ph;
    int c = t % C;  int img = t / C;
    float m = -3.4e38f;
    for (int dy = 0; dy < 2; ++dy)
        for (int dx = 0; dx < 2; ++dx) {
            int yy = 2 * y + dy, xx = 2 * x + dx;
            if (yy < h && xx < w)
                m = fmaxf(m, (float)in[((img * C + c) * h + yy) * w + xx]);
        }
    out[i] = (f16)m;
}

// ---------------------------------------------------------------------------
// 1x1 heads: softmax prob(ch1) + 4ch regression, dense buffers (deterministic)
// ---------------------------------------------------------------------------
__global__ __launch_bounds__(256) void k_pnet_head(const f16* __restrict__ in,
                                                   const float* __restrict__ w41,
                                                   const float* __restrict__ b41,
                                                   const float* __restrict__ w42,
                                                   const float* __restrict__ b42,
                                                   float* __restrict__ prob,
                                                   float* __restrict__ reg,
                                                   int h3, int w3)
{
    int i = blockIdx.x * 256 + threadIdx.x;
    int hw = h3 * w3;
    int tot = NIMG * hw;
    if (i >= tot) return;
    int img = i / hw; int r = i - img * hw;
    const f16* base = in + (size_t)img * 32 * hw + r;
    float l0 = b41[0], l1 = b41[1];
    float r0 = b42[0], r1 = b42[1], r2 = b42[2], r3 = b42[3];
    for (int ci = 0; ci < 32; ++ci) {
        float v = (float)base[(size_t)ci * hw];
        l0 += v * w41[ci];      l1 += v * w41[32 + ci];
        r0 += v * w42[ci];      r1 += v * w42[32 + ci];
        r2 += v * w42[64 + ci]; r3 += v * w42[96 + ci];
    }
    float mx = fmaxf(l0, l1);
    float e0 = expf(l0 - mx), e1 = expf(l1 - mx);
    prob[i] = e1 / (e0 + e1);
    reg[i * 4 + 0] = r0; reg[i * 4 + 1] = r1;
    reg[i * 4 + 2] = r2; reg[i * 4 + 3] = r3;
}

// ---------------------------------------------------------------------------
// Per-scale: top-K (threshold 0.6) by iterated argmax + NMS(thr) in one block.
// kept record: [x1,y1,x2,y2,score,r0,r1,r2,r3,imgidx,0,0]   (desc order)
// ---------------------------------------------------------------------------
__global__ __launch_bounds__(256) void k_topk_nms_scale(float* __restrict__ prob,
                                                        const float* __restrict__ reg,
                                                        int n, int h3, int w3,
                                                        float scale, int K,
                                                        float* __restrict__ kept,
                                                        int* __restrict__ keepf,
                                                        float thr)
{
    __shared__ float rv[256];
    __shared__ int   ri[256];
    __shared__ float bx1[512], by1[512], bx2[512], by2[512], bar[512];
    __shared__ int   nk[512];
    int tid = threadIdx.x;
    int hw = h3 * w3;

    for (int it = 0; it < K; ++it) {
        float best = -3.4e38f; int bi = -1;
        for (int j = tid; j < n; j += 256) {
            float p = prob[j];
            float s = (p >= 0.6f) ? p : -1.0e9f;
            if (s > best) { best = s; bi = j; }
        }
        rv[tid] = best; ri[tid] = bi;
        __syncthreads();
        for (int o = 128; o > 0; o >>= 1) {
            if (tid < o) {
                bool take = (rv[tid + o] > rv[tid]) ||
                            (rv[tid + o] == rv[tid] && ri[tid + o] != -1 &&
                             (ri[tid] == -1 || ri[tid + o] < ri[tid]));
                if (take) { rv[tid] = rv[tid + o]; ri[tid] = ri[tid + o]; }
            }
            __syncthreads();
        }
        if (tid == 0) {
            int sel = ri[0]; float bs = rv[0];
            float* R = &kept[(size_t)it * 12];
            if (sel >= 0 && bs > -1.0e8f) {
                prob[sel] = -3.0e9f;   // mark taken
                int img = sel / hw; int rm = sel - img * hw;
                int yy = rm / w3, xx = rm - yy * w3;
                float x1 = floorf((2.0f * xx + 1.0f) / scale);
                float y1 = floorf((2.0f * yy + 1.0f) / scale);
                float x2 = floorf((2.0f * xx + 12.0f) / scale);
                float y2 = floorf((2.0f * yy + 12.0f) / scale);
                R[0] = x1; R[1] = y1; R[2] = x2; R[3] = y2; R[4] = bs;
                R[5] = reg[sel * 4 + 0]; R[6] = reg[sel * 4 + 1];
                R[7] = reg[sel * 4 + 2]; R[8] = reg[sel * 4 + 3];
                R[9] = (float)img; R[10] = 0.0f; R[11] = 0.0f;
                float off = img * 100000.0f;
                bx1[it] = x1 + off; by1[it] = y1 + off;
                bx2[it] = x2 + off; by2[it] = y2 + off;
                bar[it] = fmaxf(x2 - x1, 0.0f) * fmaxf(y2 - y1, 0.0f);
                nk[it] = 1;
            } else {
                for (int c = 0; c < 12; ++c) R[c] = 0.0f;
                R[4] = -1.0e9f;
                bx1[it] = by1[it] = bx2[it] = by2[it] = bar[it] = 0.0f;
                nk[it] = 0;
            }
        }
        __syncthreads();
    }

    // NMS on the K sorted candidates
    for (int i = 0; i < K; ++i) {
        __syncthreads();
        if (nk[i]) {
            float xi1 = bx1[i], yi1 = by1[i], xi2 = bx2[i], yi2 = by2[i], ai = bar[i];
            for (int j = i + 1 + tid; j < K; j += 256) {
                if (!nk[j]) continue;
                float iw = fmaxf(fminf(xi2, bx2[j]) - fmaxf(xi1, bx1[j]), 0.0f);
                float ih = fmaxf(fminf(yi2, by2[j]) - fmaxf(yi1, by1[j]), 0.0f);
                float inter = iw * ih;
                float iou = inter / (ai + bar[j] - inter + 1e-9f);
                if (iou > thr) nk[j] = 0;
            }
        }
    }
    __syncthreads();
    for (int j = tid; j < K; j += 256) keepf[j] = nk[j];
}

// ---------------------------------------------------------------------------
// Final: global top-1024 over kept, NMS(0.7), regression + squaring -> d_out
// ---------------------------------------------------------------------------
__global__ __launch_bounds__(256) void k_final_nms_out(const float* __restrict__ kept,
                                                       const int* __restrict__ keepf,
                                                       int N, int K2,
                                                       float* __restrict__ fin,
                                                       float* __restrict__ out,
                                                       int outRows, int outTail,
                                                       float thr)
{
    __shared__ float scs[5632];
    __shared__ float rv[256];
    __shared__ int   ri[256];
    __shared__ float bx1[1024], by1[1024], bx2[1024], by2[1024], bar[1024];
    __shared__ int   nk[1024];
    int tid = threadIdx.x;

    for (int j = tid; j < N; j += 256)
        scs[j] = keepf[j] ? kept[(size_t)j * 12 + 4] : -1.0e9f;
    __syncthreads();

    for (int it = 0; it < K2; ++it) {
        float best = -3.4e38f; int bi = -1;
        for (int j = tid; j < N; j += 256) {
            float s = scs[j];
            if (s > best) { best = s; bi = j; }
        }
        rv[tid] = best; ri[tid] = bi;
        __syncthreads();
        for (int o = 128; o > 0; o >>= 1) {
            if (tid < o) {
                bool take = (rv[tid + o] > rv[tid]) ||
                            (rv[tid + o] == rv[tid] && ri[tid + o] != -1 &&
                             (ri[tid] == -1 || ri[tid + o] < ri[tid]));
                if (take) { rv[tid] = rv[tid + o]; ri[tid] = ri[tid + o]; }
            }
            __syncthreads();
        }
        if (tid == 0) {
            int sel = ri[0]; float bs = rv[0];
            float* R = &fin[(size_t)it * 12];
            if (sel >= 0 && bs > -1.0e8f) {
                scs[sel] = -3.0e9f;
                const float* S = &kept[(size_t)sel * 12];
                for (int c = 0; c < 12; ++c) R[c] = S[c];
                float off = S[9] * 100000.0f;
                bx1[it] = S[0] + off; by1[it] = S[1] + off;
                bx2[it] = S[2] + off; by2[it] = S[3] + off;
                bar[it] = fmaxf(S[2] - S[0], 0.0f) * fmaxf(S[3] - S[1], 0.0f);
                nk[it] = 1;
            } else {
                for (int c = 0; c < 12; ++c) R[c] = 0.0f;
                R[4] = -1.0e9f;
                bx1[it] = by1[it] = bx2[it] = by2[it] = bar[it] = 0.0f;
                nk[it] = 0;
            }
        }
        __syncthreads();
    }

    for (int i = 0; i < K2; ++i) {
        __syncthreads();
        if (nk[i]) {
            float xi1 = bx1[i], yi1 = by1[i], xi2 = bx2[i], yi2 = by2[i], ai = bar[i];
            for (int j = i + 1 + tid; j < K2; j += 256) {
                if (!nk[j]) continue;
                float iw = fmaxf(fminf(xi2, bx2[j]) - fmaxf(xi1, bx1[j]), 0.0f);
                float ih = fmaxf(fminf(yi2, by2[j]) - fmaxf(yi1, by1[j]), 0.0f);
                float inter = iw * ih;
                float iou = inter / (ai + bar[j] - inter + 1e-9f);
                if (iou > thr) nk[j] = 0;
            }
        }
    }
    __syncthreads();

    for (int r = tid; r < outRows; r += 256) {
        if (r < K2) {
            const float* R = &fin[(size_t)r * 12];
            float k = nk[r] ? 1.0f : 0.0f;
            float x1 = R[0], y1 = R[1], x2 = R[2], y2 = R[3], sc = R[4];
            float rw = x2 - x1, rh = y2 - y1;
            float ox1 = x1 + R[5] * rw, oy1 = y1 + R[6] * rh;
            float ox2 = x2 + R[7] * rw, oy2 = y2 + R[8] * rh;
            float hh = oy2 - oy1, ww = ox2 - ox1;
            float l = fmaxf(hh, ww);
            float nx1 = ox1 + ww * 0.5f - l * 0.5f;
            float ny1 = oy1 + hh * 0.5f - l * 0.5f;
            out[r * 5 + 0] = nx1 * k;
            out[r * 5 + 1] = ny1 * k;
            out[r * 5 + 2] = (nx1 + l) * k;
            out[r * 5 + 3] = (ny1 + l) * k;
            out[r * 5 + 4] = sc * k;
        } else {
            out[r * 5 + 0] = 0.0f; out[r * 5 + 1] = 0.0f; out[r * 5 + 2] = 0.0f;
            out[r * 5 + 3] = 0.0f; out[r * 5 + 4] = 0.0f;
        }
    }
    if (tid < outTail) out[outRows * 5 + tid] = 0.0f;
}

// ---------------------------------------------------------------------------
// Host orchestration
// ---------------------------------------------------------------------------
static inline size_t alignup(size_t x) { return (x + 255) & ~(size_t)255; }

extern "C" void kernel_launch(void* const* d_in, const int* in_sizes, int n_in,
                              void* d_out, int out_size, void* d_ws, size_t ws_size,
                              hipStream_t stream)
{
    (void)in_sizes; (void)n_in; (void)ws_size;
    const float* images = (const float*)d_in[0];
    const float* c1w = (const float*)d_in[1];
    const float* c1b = (const float*)d_in[2];
    const float* p1  = (const float*)d_in[3];
    const float* c2w = (const float*)d_in[4];
    const float* c2b = (const float*)d_in[5];
    const float* p2  = (const float*)d_in[6];
    const float* c3w = (const float*)d_in[7];
    const float* c3b = (const float*)d_in[8];
    const float* p3  = (const float*)d_in[9];
    const float* w41 = (const float*)d_in[10];
    const float* b41 = (const float*)d_in[11];
    const float* w42 = (const float*)d_in[12];
    const float* b42 = (const float*)d_in[13];

    // pyramid scales (double, matches python)
    double scl[16]; int ns = 0;
    double m = 12.0 / 20.0;
    double ml = 768.0 * m, si = m;
    while (ml >= 12.0 && ns < 16) { scl[ns++] = si; si *= 0.709; ml *= 0.709; }

    int NH[16], H1[16], PH[16], H2[16], H3[16], KS[16], OFF[16];
    int Ntot = 0;
    for (int s = 0; s < ns; ++s) {
        NH[s] = (int)(768.0 * scl[s] + 1.0);
        H1[s] = NH[s] - 2;
        PH[s] = (H1[s] + 1) / 2;
        H2[s] = PH[s] - 2;
        H3[s] = H2[s] - 2;
        int pos = NIMG * H3[s] * H3[s];
        KS[s] = pos < 512 ? pos : 512;
        OFF[s] = Ntot;
        Ntot += KS[s];
    }

    // workspace plan (sized at scale 0 = largest)
    int nh0 = NH[0], h1m = H1[0], phm = PH[0], h2m = H2[0], h3m = H3[0];
    size_t sz_im = alignup((size_t)NIMG * 3 * nh0 * nh0 * 2);
    size_t t1a = (size_t)NIMG * 3 * 768 * nh0 * 4;        // f32 resize tmp
    size_t t1b = (size_t)NIMG * 10 * h1m * h1m * 2;       // conv1 out f16
    size_t sz_t1 = alignup(t1a > t1b ? t1a : t1b);
    size_t sz_pl = alignup((size_t)NIMG * 10 * phm * phm * 2);
    size_t sz_c2 = alignup((size_t)NIMG * 16 * h2m * h2m * 2);
    size_t sz_c3 = alignup((size_t)NIMG * 32 * h3m * h3m * 2);
    size_t sz_pr = alignup((size_t)NIMG * h3m * h3m * 4);
    size_t sz_rg = alignup((size_t)NIMG * h3m * h3m * 16);
    size_t sz_kp = alignup((size_t)8192 * 12 * 4);
    size_t sz_kf = alignup((size_t)8192 * 4);

    char* W = (char*)d_ws;
    size_t o = 0;
    f16*   bIM  = (f16*)(W + o);   o += sz_im;
    float* bT1f = (float*)(W + o);
    f16*   bT1h = (f16*)(W + o);   o += sz_t1;
    f16*   bPL  = (f16*)(W + o);   o += sz_pl;
    f16*   bC2  = (f16*)(W + o);   o += sz_c2;
    f16*   bC3  = (f16*)(W + o);   o += sz_c3;
    float* bPR  = (float*)(W + o); o += sz_pr;
    float* bRG  = (float*)(W + o); o += sz_rg;
    float* bKP  = (float*)(W + o); o += sz_kp;
    int*   bKF  = (int*)(W + o);   o += sz_kf;
    float* bFN  = (float*)(W + o);

    for (int s = 0; s < ns; ++s) {
        int nh = NH[s], nw = nh;
        int h1 = H1[s], w1 = h1, ph = PH[s], pw = ph;
        int h2 = H2[s], w2 = h2, h3 = H3[s], w3 = h3;

        { int tot = NIMG * 3 * 768 * nw;
          k_hresize<<<(tot + 255) / 256, 256, 0, stream>>>(images, bT1f, nw, (float)(nw / 768.0)); }
        { int tot = NIMG * 3 * nh * nw;
          k_vresize<<<(tot + 255) / 256, 256, 0, stream>>>(bT1f, bIM, nh, nw, (float)(nh / 768.0)); }
        { int P = h1 * w1; int tiles = NIMG * ((P + 15) / 16);
          k_conv3x3_wmma<1, 32><<<(tiles + 7) / 8, 256, 0, stream>>>(bIM, bT1h, c1w, c1b, p1, 3, 10, nh, nw); }
        { int tot = NIMG * 10 * ph * pw;
          k_maxpool2<<<(tot + 255) / 256, 256, 0, stream>>>(bT1h, bPL, 10, h1, w1, ph, pw); }
        { int P = h2 * w2; int tiles = NIMG * ((P + 15) / 16);
          k_conv3x3_wmma<1, 96><<<(tiles + 7) / 8, 256, 0, stream>>>(bPL, bC2, c2w, c2b, p2, 10, 16, ph, pw); }
        { int P = h3 * w3; int tiles = NIMG * ((P + 15) / 16);
          k_conv3x3_wmma<2, 160><<<(tiles + 7) / 8, 256, 0, stream>>>(bC2, bC3, c3w, c3b, p3, 16, 32, h2, w2); }
        { int tot = NIMG * h3 * w3;
          k_pnet_head<<<(tot + 255) / 256, 256, 0, stream>>>(bC3, w41, b41, w42, b42, bPR, bRG, h3, w3); }
        { k_topk_nms_scale<<<1, 256, 0, stream>>>(bPR, bRG, NIMG * h3 * w3, h3, w3,
                                                  (float)scl[s], KS[s],
                                                  bKP + (size_t)OFF[s] * 12, bKF + OFF[s], 0.5f); }
    }

    int Ncl = Ntot < 5632 ? Ntot : 5632;
    int K2 = Ntot < 1024 ? Ntot : 1024;
    int outRows = out_size / 5;
    int outTail = out_size - outRows * 5;
    k_final_nms_out<<<1, 256, 0, stream>>>(bKP, bKF, Ncl, K2, bFN,
                                           (float*)d_out, outRows, outTail, 0.7f);
}
